// GCN_73504070304090
// MI455X (gfx1250) — matrix operand order repaired and checked by reading, not compiled, not run on hardware
//
#include <hip/hip_runtime.h>

// ---------------------------------------------------------------------------
// GCN on MI455X (gfx1250, wave32, WMMA bf16 16x16x32, f32 accumulate)
//
// Algebraic rewrite: adj1 @ (outer(a,b) @ W) == S @ W with
//   S[r,h] = sum_i a[i,h] * (reshape(adj1[r],[N,N]) @ b)[i,h]
// Both edge convolutions (gc_e with x_init, gc_e2's extra term with x1) are
// computed in ONE fused pass over adj1 (226 MB read once ~ 9.7us @ 23.3TB/s),
// 43.4 GFLOP of bf16 WMMA total. edge_feat (113 MB) is never materialized.
// ---------------------------------------------------------------------------

#define NN    384
#define NFEAT 512
#define NHID  192
#define NCLS  64

typedef __attribute__((ext_vector_type(16))) __bf16 v16bf;
typedef __attribute__((ext_vector_type(8)))  float  v8f;

union FragBF {
    v16bf v;
    unsigned short u[16];
};

static __device__ __forceinline__ unsigned short f2bf(float f) {
    // round-to-nearest-even f32 -> bf16 bits
    unsigned int x = __float_as_uint(f);
    x += 0x7FFFu + ((x >> 16) & 1u);
    return (unsigned short)(x >> 16);
}

// ---------------------------------------------------------------------------
// fused bilinear: for one r,
//   S1 [r,h] = sum_i x_init[i,h] * (A_r @ x_init)[i,h]
//   S2b[r,h] = sum_i x1[i,h]     * (A_r @ x1)[i,h]      A_r = adj1[r] as [N,N]
// grid = 384 blocks (one r each), block = 768 threads = 24 waves.
//   waves 0..11  : h-tile w*16 against U0 (Xb / x_init) -> S1
//   waves 12..23 : h-tile (w-12)*16 against U1 (X1b / x1) -> S2b
// A_r rows staged f32->bf16 through LDS once, consumed by both wave groups.
// B fragments preloaded per wave into VGPRs (12 chunks of K=32 = 96 VGPRs).
// ---------------------------------------------------------------------------
__global__ __launch_bounds__(768, 1) void bilinear2_kernel(
    const float* __restrict__ adj1,         // [NN, NN*NN] f32
    const unsigned short* __restrict__ U0b, // bf16(x_init) [NN, NHID]
    const float* __restrict__ U0f,          // x_init f32
    const unsigned short* __restrict__ U1b, // bf16(x1) [NN, NHID]
    const float* __restrict__ U1f,          // x1 f32
    float* __restrict__ S1,                 // [NN, NHID] f32
    float* __restrict__ S2b)                // [NN, NHID] f32
{
    const int r    = blockIdx.x;
    const int t    = threadIdx.x;
    const int lane = t & 31;
    const int wv   = t >> 5;          // 0..23
    const int grp  = wv >= 12;        // 0 -> (U0,S1), 1 -> (U1,S2b)
    const int h0   = (wv - (grp ? 12 : 0)) << 4;
    const int lhi  = lane >> 4;       // 0/1
    const int llo  = lane & 15;

    const unsigned short* Ub = grp ? U1b : U0b;
    const float*          Uf = grp ? U1f : U0f;
    float*                So = grp ? S2b : S1;

    constexpr int LDSP = NN + 8;      // padded bf16 row stride (bank spread)
    __shared__ unsigned short lds_A[16 * LDSP];

    // Preload B fragments: B[k][n] with k_local = lhi*8 + (e&7) + 16*(e>>3), n = llo
    FragBF bf[12];
    for (int jc = 0; jc < 12; ++jc) {
        const int kb = jc * 32 + (lhi << 3);
#pragma unroll
        for (int e = 0; e < 16; ++e)
            bf[jc].u[e] = Ub[(kb + (e & 7) + ((e >> 3) << 4)) * NHID + h0 + llo];
    }

    // LDS fill mapping: 768 threads x 8 consecutive f32 = 16x384 tile
    const int fr = t / 48;            // row 0..15
    const int fc = (t % 48) * 8;      // col 0..376

    float accv = 0.f;

    for (int it = 0; it < 24; ++it) {
        const int i0 = it << 4;
        // stage A_r[i0..i0+15][0..383] into LDS as bf16
        const float* src = adj1 + (size_t)r * (NN * NN) + (size_t)(i0 + fr) * NN + fc;
        unsigned short* dst = lds_A + fr * LDSP + fc;
#pragma unroll
        for (int q = 0; q < 8; ++q) dst[q] = f2bf(src[q]);
        __syncthreads();

        v8f c = {};
#pragma unroll
        for (int jc = 0; jc < 12; ++jc) {
            FragBF a;
            // A frag: m = llo, k = jc*32 + lhi*8 + e (e<8), +16 for e>=8
            const unsigned short* arow = lds_A + llo * LDSP + jc * 32 + (lhi << 3);
#pragma unroll
            for (int e = 0; e < 8; ++e) {
                a.u[e]     = arow[e];
                a.u[8 + e] = arow[16 + e];
            }
            c = __builtin_amdgcn_wmma_f32_16x16x32_bf16(
                    false, a.v, false, bf[jc].v, (short)0, c, false, false);
        }
        // weighted reduce over this i-tile: D(M,N) -> M = i0 + v + 8*lhi, N = llo
#pragma unroll
        for (int v = 0; v < 8; ++v) {
            const int mi = i0 + v + (lhi << 3);
            accv += c[v] * Uf[mi * NHID + h0 + llo];
        }
        __syncthreads();
    }

    accv += __shfl_xor(accv, 16);
    if (lhi == 0) So[r * NHID + h0 + llo] = accv;
}

// ---------------------------------------------------------------------------
// generic WMMA GEMM: C = act( Aeff @ B + bias ), Aeff = a_act(A + Aa1 + Aa2)
// f32 in, bf16 multiplicands, f32 accumulate; optional bf16 mirror of C.
// One wave per 16x16 output tile; M,N mult of 16; K mult of 32.
// ---------------------------------------------------------------------------
__global__ __launch_bounds__(128) void gemm_wmma(
    const float* __restrict__ A,  const float* __restrict__ Aa1,
    const float* __restrict__ Aa2, int a_relu,
    const float* __restrict__ B,  const float* __restrict__ bias,
    float* __restrict__ C, unsigned short* __restrict__ Cbf, int c_relu,
    int M, int K, int N)
{
    const int wid  = (int)((blockIdx.x * blockDim.x + threadIdx.x) >> 5);
    const int lane = threadIdx.x & 31;
    const int nt   = N >> 4;
    const int T    = (M >> 4) * nt;
    if (wid >= T) return;                 // wave-uniform: EXEC stays full

    const int m0  = (wid / nt) << 4;
    const int n0  = (wid % nt) << 4;
    const int lhi = lane >> 4;
    const int llo = lane & 15;

    v8f c = {};
    for (int k0 = 0; k0 < K; k0 += 32) {
        FragBF a, b;
        const int am = m0 + llo;
#pragma unroll
        for (int e = 0; e < 16; ++e) {
            const int ka = k0 + (lhi << 4) + e;
            float av = A[am * K + ka];
            if (Aa1) av += Aa1[am * K + ka];
            if (Aa2) av += Aa2[am * K + ka];
            if (a_relu) av = fmaxf(av, 0.f);
            a.u[e] = f2bf(av);
            const int kb = k0 + (lhi << 4) + e;
            b.u[e] = f2bf(B[kb * N + n0 + llo]);
        }
        c = __builtin_amdgcn_wmma_f32_16x16x32_bf16(
                false, a.v, false, b.v, (short)0, c, false, false);
    }
#pragma unroll
    for (int v = 0; v < 8; ++v) {
        const int mi = m0 + v + (lhi << 3);
        const int ni = n0 + llo;
        float val = c[v] + (bias ? bias[ni] : 0.f);
        if (c_relu) val = fmaxf(val, 0.f);
        C[mi * N + ni] = val;
        if (Cbf) Cbf[mi * N + ni] = f2bf(val);
    }
}

// ---------------------------------------------------------------------------
// row-wise log_softmax over [NN, 64]: one wave per row, 2 cols/lane, in-place.
// ---------------------------------------------------------------------------
__global__ __launch_bounds__(32) void lsm_kernel(float* __restrict__ io)
{
    const int row  = blockIdx.x;
    const int lane = threadIdx.x;
    float v0 = io[row * NCLS + lane];
    float v1 = io[row * NCLS + 32 + lane];
    float m = fmaxf(v0, v1);
#pragma unroll
    for (int o = 16; o; o >>= 1) m = fmaxf(m, __shfl_xor(m, o));
    float s = __expf(v0 - m) + __expf(v1 - m);
#pragma unroll
    for (int o = 16; o; o >>= 1) s += __shfl_xor(s, o);
    const float lz = m + __logf(s);
    io[row * NCLS + lane]      = v0 - lz;
    io[row * NCLS + 32 + lane] = v1 - lz;
}

// ---------------------------------------------------------------------------

extern "C" void kernel_launch(void* const* d_in, const int* in_sizes, int n_in,
                              void* d_out, int out_size, void* d_ws, size_t ws_size,
                              hipStream_t stream) {
    const float* x     = (const float*)d_in[0];
    const float* adj   = (const float*)d_in[1];
    const float* adj1  = (const float*)d_in[2];
    const float* W_emb = (const float*)d_in[3];
    const float* b_emb = (const float*)d_in[4];
    const float* W_cls = (const float*)d_in[5];
    const float* b_cls = (const float*)d_in[6];
    const float* W1    = (const float*)d_in[7];
    const float* b1    = (const float*)d_in[8];
    const float* W2    = (const float*)d_in[9];
    const float* b2    = (const float*)d_in[10];
    const float* We    = (const float*)d_in[11];
    const float* be    = (const float*)d_in[12];
    const float* We2   = (const float*)d_in[13];
    const float* be2   = (const float*)d_in[14];

    // workspace partition (256B aligned)
    char* p = (char*)d_ws;
    auto alloc = [&](size_t bytes) -> void* {
        void* q = (void*)p;
        p += (bytes + 255) & ~(size_t)255;
        return q;
    };
    const size_t MATF = (size_t)NN * NHID * sizeof(float);
    const size_t MATB = (size_t)NN * NHID * sizeof(unsigned short);
    float*          x_init = (float*)alloc(MATF);
    unsigned short* Xb     = (unsigned short*)alloc(MATB);
    float*          t1     = (float*)alloc(MATF);   // reused as t2
    float*          x1     = (float*)alloc(MATF);
    unsigned short* X1b    = (unsigned short*)alloc(MATB);
    float*          S1     = (float*)alloc(MATF);
    float*          S2b    = (float*)alloc(MATF);
    float*          x_e    = (float*)alloc(MATF);
    float*          x_mid  = (float*)alloc(MATF);
    float*          x_e2   = (float*)alloc(MATF);

    auto gemm = [&](const float* A, const float* Aa1, const float* Aa2, int arelu,
                    const float* B, const float* bias, float* C, unsigned short* Cb,
                    int crelu, int M, int K, int N) {
        const int T = (M / 16) * (N / 16);
        const int blocks = (T + 3) / 4;
        gemm_wmma<<<blocks, 128, 0, stream>>>(A, Aa1, Aa2, arelu, B, bias,
                                              C, Cb, crelu, M, K, N);
    };

    // 1) x_init = x @ W_emb + b_emb         (also bf16 mirror Xb)
    gemm(x, nullptr, nullptr, 0, W_emb, b_emb, x_init, Xb, 0, NN, NFEAT, NHID);
    // 2) t1 = x_init @ W1
    gemm(x_init, nullptr, nullptr, 0, W1, nullptr, t1, nullptr, 0, NN, NHID, NHID);
    // 3) x1 = relu(adj @ t1 + b1)           (also bf16 mirror X1b)
    gemm(adj, nullptr, nullptr, 0, t1, b1, x1, X1b, 1, NN, NN, NHID);
    // 4+5) fused single pass over adj1:
    //      S1[r,h]  = sum_i x_init[i,h] * (adj1_r @ x_init)[i,h]
    //      S2b[r,h] = sum_i x1[i,h]     * (adj1_r @ x1)[i,h]
    bilinear2_kernel<<<NN, 768, 0, stream>>>(adj1, Xb, x_init, X1b, x1, S1, S2b);
    // 6) x_e = relu(S1 @ We + be)           (== relu(adj1 @ (edge_feat@We) + be))
    gemm(S1, nullptr, nullptr, 0, We, be, x_e, nullptr, 1, NN, NHID, NHID);
    // 7) t2 = (x1 + x_e + x_init) @ W2      (xs folded into A-side)
    gemm(x1, x_e, x_init, 0, W2, nullptr, t1, nullptr, 0, NN, NHID, NHID);
    // 8) x_mid = adj @ t2 + b2
    gemm(adj, nullptr, nullptr, 0, t1, b2, x_mid, nullptr, 0, NN, NN, NHID);
    // 9) x_e2 = (S1 + S2b) @ We2 + be2      (== adj1 @ ((ef+ef1)@We2) + be2)
    gemm(S1, S2b, nullptr, 0, We2, be2, x_e2, nullptr, 0, NN, NHID, NHID);
    // 10) logits = relu(x_mid + x_e2) @ W_cls + b_cls   -> d_out
    gemm(x_mid, x_e2, nullptr, 1, W_cls, b_cls, (float*)d_out, nullptr, 0,
         NN, NHID, NCLS);
    // 11) in-place row log_softmax
    lsm_kernel<<<NN, 32, 0, stream>>>((float*)d_out);
}
